// GraphSage2_67551245631650
// MI455X (gfx1250) — matrix operand order repaired and checked by reading, not compiled, not run on hardware
//
#include <hip/hip_runtime.h>
#include <hip/hip_bf16.h>
#include <math.h>

#define N_NODES 100000
#define N_EDGES 3200000
#define IN_F    128
#define HID     16

typedef __attribute__((ext_vector_type(16))) _Float16 v16h;
typedef __attribute__((ext_vector_type(8)))  _Float16 v8h;
typedef __attribute__((ext_vector_type(8)))  float    v8f;
typedef __attribute__((ext_vector_type(4)))  float    v4f;

// ---------------- workspace layout (bytes) ----------------
static constexpr size_t al256(size_t x) { return (x + 255) & ~(size_t)255; }
static constexpr size_t O_BPK = 0;                               // 11 blocks * 32 lanes * 16 halves
static constexpr size_t SZ_BPK = 11ull * 32 * 16 * 2;            // 11264 B
static constexpr size_t O_CNT = al256(O_BPK + SZ_BPK);           // N u32
static constexpr size_t O_P1  = al256(O_CNT + 4ull * N_NODES);   // N*16 f32
static constexpr size_t O_R1  = al256(O_P1  + 64ull * N_NODES);
static constexpr size_t O_AG1 = al256(O_R1  + 64ull * N_NODES);
static constexpr size_t O_H1P = al256(O_AG1 + 64ull * N_NODES);  // N*32 f32 (zero-padded)
static constexpr size_t O_P2  = al256(O_H1P + 128ull * N_NODES);
static constexpr size_t O_R2  = al256(O_P2  + 64ull * N_NODES);
static constexpr size_t O_AG2 = al256(O_R2  + 64ull * N_NODES);
static constexpr size_t O_H2H = al256(O_AG2 + 64ull * N_NODES);  // N*16 f16

// ---------------- kernels ----------------

__global__ void k_zero(float* __restrict__ agg1, float* __restrict__ agg2,
                       unsigned* __restrict__ cntU) {
  int tid = blockIdx.x * blockDim.x + threadIdx.x;
  if (tid < N_NODES * HID) { agg1[tid] = 0.f; agg2[tid] = 0.f; }
  if (tid < N_NODES) cntU[tid] = 0u;
}

__global__ void k_degree(const int* __restrict__ dst, unsigned* __restrict__ cntU) {
  int e = blockIdx.x * blockDim.x + threadIdx.x;
  if (e < N_EDGES) atomicAdd(&cntU[dst[e]], 1u);
}

// Pre-swizzle weights into CDNA5 16-bit B-matrix lane layout (f16).
// B-packed block b, lane L, slot j holds W[K][n] with n=L&15, K=8*(L>>4)+j (j<8), 8+8*(L>>4)+j (j>=8).
// blocks 0..7: w1 chunks (kc=b>>1, tile b&1: 0=w1_l, 1=w1_r), K global = kc*32+kloc
// blocks 8,9 : w2_l / w2_r padded to K=32 (rows 16..31 zero)
// block 10   : fc1_w (32x16)
__global__ void k_prepw(const float* __restrict__ w1l, const float* __restrict__ w1r,
                        const float* __restrict__ w2l, const float* __restrict__ w2r,
                        const float* __restrict__ fc1w, _Float16* __restrict__ bpk) {
  int idx = blockIdx.x * blockDim.x + threadIdx.x;
  if (idx >= 11 * 32 * 16) return;
  int j    = idx & 15;
  int lane = (idx >> 4) & 31;
  int blk  = idx >> 9;
  int n    = lane & 15;
  int kloc = (j < 8) ? ((lane >> 4) << 3) + j : 16 + ((lane >> 4) << 3) + (j - 8);  // 0..31 within a K-chunk
  float v;
  if (blk < 8) {
    int kc = blk >> 1;
    const float* W = (blk & 1) ? w1r : w1l;
    v = W[(kc * 32 + kloc) * HID + n];
  } else if (blk == 8) {
    v = (kloc < HID) ? w2l[kloc * HID + n] : 0.f;
  } else if (blk == 9) {
    v = (kloc < HID) ? w2r[kloc * HID + n] : 0.f;
  } else {
    v = fc1w[kloc * HID + n];
  }
  bpk[idx] = (_Float16)v;
}

// Dense projection GEMM: out{P,R}[N x 16] = in[N x (nchunks*32)] @ B{tile0,tile1}
// One wave handles 16 rows. A built in the ISA 16-bit 16x32 layout on the fly.
__global__ void k_proj(const float* __restrict__ in, int in_stride,
                       const _Float16* __restrict__ bpk, int nchunks,
                       float* __restrict__ outP, float* __restrict__ outR) {
  int wave = (blockIdx.x * blockDim.x + threadIdx.x) >> 5;
  int rowBase = wave * 16;
  if (rowBase >= N_NODES) return;          // wave-uniform: EXEC stays all-ones
  int lane = threadIdx.x & 31;
  int half = lane >> 4;
  int r    = lane & 15;
  const float* rowp = in + (size_t)(rowBase + r) * in_stride;

  v8f accP = {}; v8f accR = {};
  for (int kc = 0; kc < nchunks; ++kc) {
    const float* ap = rowp + kc * 32 + half * 8;
    v4f q0 = *(const v4f*)(ap);
    v4f q1 = *(const v4f*)(ap + 4);
    v4f q2 = *(const v4f*)(ap + 16);
    v4f q3 = *(const v4f*)(ap + 20);
    v16h a;
#pragma unroll
    for (int j = 0; j < 4; ++j) {
      a[j]      = (_Float16)q0[j];
      a[4 + j]  = (_Float16)q1[j];
      a[8 + j]  = (_Float16)q2[j];
      a[12 + j] = (_Float16)q3[j];
    }
    v16h bP = *(const v16h*)(bpk + ((size_t)(kc * 2 + 0) * 32 + lane) * 16);
    v16h bR = *(const v16h*)(bpk + ((size_t)(kc * 2 + 1) * 32 + lane) * 16);
    accP = __builtin_amdgcn_wmma_f32_16x16x32_f16(false, a, false, bP, (short)0, accP, false, false);
    accR = __builtin_amdgcn_wmma_f32_16x16x32_f16(false, a, false, bR, (short)0, accR, false, false);
  }
  // C/D layout: VGPR v, lanes 0-15 -> row v, lanes 16-31 -> row v+8; col = r
#pragma unroll
  for (int v = 0; v < 8; ++v) {
    size_t m = (size_t)(rowBase + v + 8 * half);
    outP[m * HID + r] = accP[v];
    outR[m * HID + r] = accR[v];
  }
}

// Scatter-add projected features: agg[dst] += p[src], one thread per (edge, feature)
__global__ void k_scatter(const float* __restrict__ p, const int* __restrict__ src,
                          const int* __restrict__ dst, float* __restrict__ agg) {
  long long tid = (long long)blockIdx.x * blockDim.x + threadIdx.x;
  if (tid >= (long long)N_EDGES * HID) return;
  int e = (int)(tid >> 4);
  int f = (int)(tid & 15);
  float v = p[(size_t)src[e] * HID + f];
  atomicAdd(&agg[(size_t)dst[e] * HID + f], v);
}

// h1 = relu(agg/max(cnt,1) + b + r), written zero-padded to width 32 for layer-2 GEMM
__global__ void k_fin1(const float* __restrict__ agg, const unsigned* __restrict__ cntU,
                       const float* __restrict__ rproj, const float* __restrict__ bias,
                       float* __restrict__ h1p) {
  int tid = blockIdx.x * blockDim.x + threadIdx.x;
  if (tid >= N_NODES * HID) return;
  int n = tid >> 4, f = tid & 15;
  float c = fmaxf((float)cntU[n], 1.f);
  float h = fmaxf(agg[tid] / c + bias[f] + rproj[tid], 0.f);
  h1p[(size_t)n * 32 + f] = h;
  h1p[(size_t)n * 32 + 16 + f] = 0.f;
}

// h2 = relu(agg/max(cnt,1) + b + r), stored as f16 for the edge-MLP WMMA
__global__ void k_fin2(const float* __restrict__ agg, const unsigned* __restrict__ cntU,
                       const float* __restrict__ rproj, const float* __restrict__ bias,
                       _Float16* __restrict__ h2h) {
  int tid = blockIdx.x * blockDim.x + threadIdx.x;
  if (tid >= N_NODES * HID) return;
  int n = tid >> 4, f = tid & 15;
  float c = fmaxf((float)cntU[n], 1.f);
  float h = fmaxf(agg[tid] / c + bias[f] + rproj[tid], 0.f);
  h2h[tid] = (_Float16)h;
}

// Edge MLP: one wave = 16 edges. A = [h2[src] | h2[dst]] (16x32 f16) built directly
// in the WMMA A layout, fc1 via one v_wmma_f32_16x16x32_f16.
// 2-class log_softmax needs only d = z0 - z1 per row:
//   out0 = -softplus(-d), out1 = -softplus(d)
// so reduce a single scalar per row via a reduce-scatter butterfly:
// 8 shuffles total (was 64). After the butterfly, lane r owns row (r>>1)&7.
__global__ void k_edge(const int* __restrict__ src, const int* __restrict__ dst,
                       const _Float16* __restrict__ h2h, const _Float16* __restrict__ bpkFc1,
                       const float* __restrict__ fc1b, const float* __restrict__ fc2w,
                       const float* __restrict__ fc2b, float* __restrict__ out) {
  int tile = (blockIdx.x * blockDim.x + threadIdx.x) >> 5;   // 200000 tiles exactly
  int lane = threadIdx.x & 31;
  int half = lane >> 4;
  int r    = lane & 15;
  int e    = tile * 16 + r;
  int s = src[e], d = dst[e];

  v8h hs = *(const v8h*)(h2h + (size_t)s * HID + half * 8);
  v8h hd = *(const v8h*)(h2h + (size_t)d * HID + half * 8);
  v16h a;
#pragma unroll
  for (int j = 0; j < 8; ++j) { a[j] = hs[j]; a[8 + j] = hd[j]; }
  v16h b = *(const v16h*)(bpkFc1 + (size_t)lane * 16);
  v8f c = {};
  c = __builtin_amdgcn_wmma_f32_16x16x32_f16(false, a, false, b, (short)0, c, false, false);

  float bias = fc1b[r];                       // this lane's column n = r
  float wd = fc2w[r * 2 + 0] - fc2w[r * 2 + 1];
  float db = fc2b[0] - fc2b[1];

  // per-lane partial of d = (z0 - z1) for each of the 8 rows this lane touches
  float p[8];
#pragma unroll
  for (int v = 0; v < 8; ++v) p[v] = fmaxf(c[v] + bias, 0.f) * wd;

  // reduce-scatter butterfly across the 16 lanes of this half
  bool bit3 = (r & 8) != 0;
  float k0 = bit3 ? p[4] : p[0], s0 = bit3 ? p[0] : p[4];
  float k1 = bit3 ? p[5] : p[1], s1 = bit3 ? p[1] : p[5];
  float k2 = bit3 ? p[6] : p[2], s2 = bit3 ? p[2] : p[6];
  float k3 = bit3 ? p[7] : p[3], s3 = bit3 ? p[3] : p[7];
  k0 += __shfl_xor(s0, 8, 16);
  k1 += __shfl_xor(s1, 8, 16);
  k2 += __shfl_xor(s2, 8, 16);
  k3 += __shfl_xor(s3, 8, 16);

  bool bit2 = (r & 4) != 0;
  float m0 = bit2 ? k2 : k0, t0 = bit2 ? k0 : k2;
  float m1 = bit2 ? k3 : k1, t1 = bit2 ? k1 : k3;
  m0 += __shfl_xor(t0, 4, 16);
  m1 += __shfl_xor(t1, 4, 16);

  bool bit1 = (r & 2) != 0;
  float u = bit1 ? m1 : m0, w = bit1 ? m0 : m1;
  u += __shfl_xor(w, 2, 16);
  u += __shfl_xor(u, 1, 16);                  // full 16-lane sum; row = (r>>1)&7

  float dd = u + db;                          // z0 - z1
  float l  = log1pf(expf(-fabsf(dd)));        // stable softplus core
  float o0 = (dd >= 0.f) ? -l : dd - l;       // = -softplus(-dd)
  float o1 = (dd >= 0.f) ? -dd - l : -l;      // = -softplus(dd)

  if ((r & 1) == 0) {
    int erow = tile * 16 + ((r >> 1) & 7) + 8 * half;
    float2 o; o.x = o0; o.y = o1;
    *(float2*)(out + (size_t)2 * erow) = o;
  }
}

// ---------------- launch ----------------
extern "C" void kernel_launch(void* const* d_in, const int* in_sizes, int n_in,
                              void* d_out, int out_size, void* d_ws, size_t ws_size,
                              hipStream_t stream) {
  const float* x     = (const float*)d_in[0];
  const int*   eidx  = (const int*)d_in[1];
  const float* w1_l  = (const float*)d_in[2];
  const float* b1_l  = (const float*)d_in[3];
  const float* w1_r  = (const float*)d_in[4];
  const float* w2_l  = (const float*)d_in[5];
  const float* b2_l  = (const float*)d_in[6];
  const float* w2_r  = (const float*)d_in[7];
  const float* fc1_w = (const float*)d_in[8];
  const float* fc1_b = (const float*)d_in[9];
  const float* fc2_w = (const float*)d_in[10];
  const float* fc2_b = (const float*)d_in[11];
  const int* src = eidx;
  const int* dst = eidx + N_EDGES;

  char* ws = (char*)d_ws;
  _Float16* bpk  = (_Float16*)(ws + O_BPK);
  unsigned* cntU = (unsigned*)(ws + O_CNT);
  float* p1   = (float*)(ws + O_P1);
  float* r1   = (float*)(ws + O_R1);
  float* agg1 = (float*)(ws + O_AG1);
  float* h1p  = (float*)(ws + O_H1P);
  float* p2   = (float*)(ws + O_P2);
  float* r2   = (float*)(ws + O_R2);
  float* agg2 = (float*)(ws + O_AG2);
  _Float16* h2h = (_Float16*)(ws + O_H2H);
  float* out = (float*)d_out;

  const int B = 256;
  k_zero<<<(N_NODES * HID + B - 1) / B, B, 0, stream>>>(agg1, agg2, cntU);
  k_degree<<<(N_EDGES + B - 1) / B, B, 0, stream>>>(dst, cntU);
  k_prepw<<<(11 * 32 * 16 + B - 1) / B, B, 0, stream>>>(w1_l, w1_r, w2_l, w2_r, fc1_w, bpk);

  // Layer 1: project then aggregate (8x less gather traffic than aggregating x)
  int projBlocks = ((N_NODES / 16) + 7) / 8;  // 8 waves per block
  k_proj<<<projBlocks, B, 0, stream>>>(x, IN_F, bpk, IN_F / 32, p1, r1);
  k_scatter<<<(int)(((long long)N_EDGES * HID + B - 1) / B), B, 0, stream>>>(p1, src, dst, agg1);
  k_fin1<<<(N_NODES * HID + B - 1) / B, B, 0, stream>>>(agg1, cntU, r1, b1_l, h1p);

  // Layer 2 (K padded to 32; pad rows/cols are zero)
  k_proj<<<projBlocks, B, 0, stream>>>(h1p, 32, bpk + 8ull * 512, 1, p2, r2);
  k_scatter<<<(int)(((long long)N_EDGES * HID + B - 1) / B), B, 0, stream>>>(p2, src, dst, agg2);
  k_fin2<<<(N_NODES * HID + B - 1) / B, B, 0, stream>>>(agg2, cntU, r2, b2_l, h2h);

  // Edge MLP: 200000 tiles, 8 waves/block -> 25000 blocks exactly
  k_edge<<<(N_EDGES / 16) / 8, B, 0, stream>>>(src, dst, h2h, bpk + 10ull * 512,
                                               fc1_b, fc2_w, fc2_b, out);
}
